// share_qureySA_39444979646754
// MI455X (gfx1250) — matrix operand order repaired and checked by reading, not compiled, not run on hardware
//
#include <hip/hip_runtime.h>
#include <hip/hip_bf16.h>

typedef __attribute__((ext_vector_type(16))) __bf16 v16bf;
typedef __attribute__((ext_vector_type(8)))  __bf16 v8bf;
typedef __attribute__((ext_vector_type(8)))  float  v8f;
typedef __attribute__((ext_vector_type(4)))  unsigned int v4u;
typedef __attribute__((ext_vector_type(8)))  int v8i;
typedef __attribute__((ext_vector_type(4)))  int v4i;

#define B_   8
#define S_   2048
#define C_   512
#define NTOK (B_ * S_)

// ---------- helpers ----------

// A-matrix per-lane register block (16 bf16). k order used here (B loads use the same):
// lane L holds row M = L&15; k-run [kk+16*(L>>4) .. +15] (contiguous).
__device__ inline v16bf pack_a16(const __bf16* p) {
  v8bf lo = *(const v8bf*)p;
  v8bf hi = *(const v8bf*)(p + 8);
  v16bf a;
#pragma unroll
  for (int i = 0; i < 8; ++i) { a[i] = lo[i]; a[i + 8] = hi[i]; }
  return a;
}

// RA/RB = ISA reuse-A/reuse-B hints (immediates via template params).
template <bool RA, bool RB>
__device__ inline v8f wmma_bf16(v16bf a, v16bf b, v8f c) {
  return __builtin_amdgcn_wmma_f32_16x16x32_bf16(false, a, false, b,
                                                 (short)0, c, RA, RB);
}

// ---------- kernel 1: fp32 -> bf16 convert ----------

__global__ void cvt_f32_bf16_kernel(const float* __restrict__ src,
                                    unsigned short* __restrict__ dst_u, int n) {
  __bf16* dst = (__bf16*)dst_u;
  int i = blockIdx.x * blockDim.x + threadIdx.x;
  int stride = gridDim.x * blockDim.x;
  for (; i < n; i += stride) dst[i] = (__bf16)src[i];
}

// ---------- kernel 2: fused QKV projection (out = x @ W^T + b) ----------
// 32x32 output block per wave (2x2 tiles of 16x16) -> 4 WMMAs per A/B load pair.
// wave id = mat*8192 + mt*16 + nt  (mat: 0=Q,1=K,2=V; mt: 512 row blocks; nt: 16 col blocks)

__global__ __launch_bounds__(256) void qkv_proj_kernel(
    const unsigned short* __restrict__ xb_u,
    const unsigned short* __restrict__ wq_u, const unsigned short* __restrict__ wk_u,
    const unsigned short* __restrict__ wv_u,
    const float* __restrict__ bq, const float* __restrict__ bk,
    const float* __restrict__ bv,
    unsigned short* __restrict__ q_u, unsigned short* __restrict__ k_u,
    unsigned short* __restrict__ vt_u) {
  const __bf16* xb = (const __bf16*)xb_u;
  const int wave = (blockIdx.x * 256 + (int)threadIdx.x) >> 5;
  const int lane = threadIdx.x & 31;
  const int nt  = wave & 15;
  const int mt  = (wave >> 4) & 511;
  const int mat = wave >> 13;

  const __bf16* W  = (const __bf16*)(mat == 0 ? wq_u : (mat == 1 ? wk_u : wv_u));
  const float* bias = (mat == 0 ? bq : (mat == 1 ? bk : bv));

  const int m0 = mt * 32, n0 = nt * 32;
  const int lrow = lane & 15, lhi = lane >> 4;

  v8f acc00 = {}, acc01 = {}, acc10 = {}, acc11 = {};
  const __bf16* arow0 = xb + (size_t)(m0 + lrow) * C_;
  const __bf16* arow1 = arow0 + (size_t)16 * C_;
  const __bf16* brow0 = W + (size_t)(n0 + lrow) * C_;
  const __bf16* brow1 = brow0 + (size_t)16 * C_;
#pragma unroll 2
  for (int kk = 0; kk < C_; kk += 32) {
    v16bf a0 = pack_a16(arow0 + kk + 16 * lhi);
    v16bf a1 = pack_a16(arow1 + kk + 16 * lhi);
    v16bf b0 = *(const v16bf*)(brow0 + kk + 16 * lhi);
    v16bf b1 = *(const v16bf*)(brow1 + kk + 16 * lhi);
    acc00 = wmma_bf16<false, false>(a0, b0, acc00);
    acc01 = wmma_bf16<true,  false>(a0, b1, acc01);   // reuse A0
    acc11 = wmma_bf16<false, true >(a1, b1, acc11);   // reuse B1
    acc10 = wmma_bf16<true,  false>(a1, b0, acc10);   // reuse A1
  }

  __bf16* Qb = (__bf16*)q_u;
  __bf16* Kb = (__bf16*)k_u;
  __bf16* Vt = (__bf16*)vt_u;
  const float bn0 = bias[n0 + lrow];
  const float bn1 = bias[n0 + 16 + lrow];
#pragma unroll
  for (int half = 0; half < 2; ++half) {
    v8f aL = half ? acc10 : acc00;
    v8f aR = half ? acc11 : acc01;
    const int mrow0 = m0 + 16 * half;
#pragma unroll
    for (int r = 0; r < 8; ++r) {
      int m = mrow0 + r + 8 * lhi;
      __bf16 vL = (__bf16)(aL[r] + bn0);
      __bf16 vR = (__bf16)(aR[r] + bn1);
      int colL = n0 + lrow, colR = n0 + 16 + lrow;
      if (mat == 0) {
        Qb[(size_t)m * C_ + colL] = vL;
        Qb[(size_t)m * C_ + colR] = vR;
      } else if (mat == 1) {
        Kb[(size_t)m * C_ + colL] = vL;
        Kb[(size_t)m * C_ + colR] = vR;
      } else {
        int bb = m / S_, s = m & (S_ - 1);
        size_t base = (size_t)bb * C_ * S_;
        Vt[base + (size_t)colL * S_ + s] = vL;   // V transposed
        Vt[base + (size_t)colR * S_ + s] = vR;
      }
    }
  }
}

// ---------- kernel 3: fused attention + residual ----------
// One workgroup (256 thr = 8 waves) per (batch, 16 query rows).
// LDS: scores f32 [16][2048] | P bf16 [16][2048] | Qtile bf16 [16][512]

#define SMEM_S  (16 * S_ * 4)
#define SMEM_P  (16 * S_ * 2)
#define SMEM_Q  (16 * C_ * 2)
#define SMEM_SZ (SMEM_S + SMEM_P + SMEM_Q)

__global__ __launch_bounds__(256) void attn_kernel(
    const unsigned short* __restrict__ q_u, const unsigned short* __restrict__ k_u,
    const unsigned short* __restrict__ vt_u,
    const float* __restrict__ x, float* __restrict__ out) {
  extern __shared__ char smem[];
  float*  Srow  = (float*)smem;
  __bf16* Ptile = (__bf16*)(smem + SMEM_S);
  __bf16* Qtile = (__bf16*)(smem + SMEM_S + SMEM_P);

  const __bf16* Qb = (const __bf16*)q_u;
  const __bf16* Kb = (const __bf16*)k_u;
  const __bf16* Vt = (const __bf16*)vt_u;

  const int b   = blockIdx.y;
  const int q0  = blockIdx.x * 16;
  const int tid = threadIdx.x;
  const int wave = tid >> 5, lane = tid & 31;
  const int lrow = lane & 15, lhi = lane >> 4;
  const size_t tokbase = (size_t)b * S_ + q0;

  // phase 0: stage Q tile (16 rows x 512 bf16) in LDS via the Tensor Data Mover
#if __has_builtin(__builtin_amdgcn_tensor_load_to_lds) && \
    __has_builtin(__builtin_amdgcn_s_wait_tensorcnt)
  if (wave == 0) {
    unsigned long long ga = (unsigned long long)(size_t)(Qb + tokbase * C_);
    unsigned int lds_addr = (unsigned int)(size_t)(void*)Qtile;  // LDS aperture low bits
    v4u g0;
    g0[0] = 1u;                                        // count=1, user descriptor
    g0[1] = lds_addr;                                  // lds_addr [63:32]
    g0[2] = (unsigned int)(ga & 0xffffffffu);          // global_addr lo
    g0[3] = (unsigned int)((ga >> 32) & 0x01ffffffu)   // global_addr hi (57-bit)
            | (2u << 30);                              // type=2 ("image")
    v8i g1;
    g1[0] = (1 << 16);                 // data_size=1 -> 2-byte elements (bf16)
    g1[1] = (int)((unsigned)C_ << 16); // tensor_dim0 lo16 at [63:48]
    g1[2] = (int)(16u << 16);          // tensor_dim0 hi16=0 | tensor_dim1 lo16=16
    g1[3] = (int)((unsigned)C_ << 16); // tensor_dim1 hi16=0 | tile_dim0=512
    g1[4] = 16;                        // tile_dim1=16, tile_dim2=0
    g1[5] = C_;                        // tensor_dim0_stride lo32 = 512
    g1[6] = 0;                         // stride hi16 | tensor_dim1_stride lo16
    g1[7] = 0;
    v4i gz4 = {0, 0, 0, 0};            // 2-D tensor: groups 2/3 unused
    v8i gz8 = {0, 0, 0, 0, 0, 0, 0, 0};
    __builtin_amdgcn_tensor_load_to_lds(g0, g1, gz4, gz4, gz8, 0);
    __builtin_amdgcn_s_wait_tensorcnt(0);
  }
#else
  for (int idx = tid; idx < (16 * C_) / 8; idx += 256) {
    int row = idx >> 6;
    int off = (idx & 63) * 8;
    *(v8bf*)(Qtile + row * C_ + off) =
        *(const v8bf*)(Qb + (tokbase + row) * C_ + off);
  }
#endif
  __syncthreads();

  // phase 1: scores S = Q * K^T ; wave handles pairs of key tiles (A shared)
  for (int jp = wave; jp < S_ / 32; jp += 8) {
    const int j0 = jp * 32;
    const __bf16* krow0 = Kb + ((size_t)b * S_ + j0 + lrow) * C_;
    const __bf16* krow1 = krow0 + (size_t)16 * C_;
    __builtin_prefetch(krow0 + 16 * C_, 0, 1);
    v8f acc0 = {}, acc1 = {};
#pragma unroll 2
    for (int kk = 0; kk < C_; kk += 32) {
      v16bf a = pack_a16(Qtile + lrow * C_ + kk + 16 * lhi);
      v16bf bm0 = *(const v16bf*)(krow0 + kk + 16 * lhi);
      v16bf bm1 = *(const v16bf*)(krow1 + kk + 16 * lhi);
      acc0 = wmma_bf16<false, false>(a, bm0, acc0);
      acc1 = wmma_bf16<true,  false>(a, bm1, acc1);    // reuse A
    }
#pragma unroll
    for (int r = 0; r < 8; ++r) {
      Srow[(r + 8 * lhi) * S_ + j0 + lrow]      = acc0[r];
      Srow[(r + 8 * lhi) * S_ + j0 + 16 + lrow] = acc1[r];
    }
  }
  __syncthreads();

  // phase 2: row softmax over 2048 keys; 16 threads per row
  {
    const int row = tid >> 4, sub = tid & 15;
    float mx = -3.402823466e38f;
    for (int c = sub; c < S_; c += 16) mx = fmaxf(mx, Srow[row * S_ + c]);
#pragma unroll
    for (int off = 8; off; off >>= 1) mx = fmaxf(mx, __shfl_xor(mx, off, 32));
    float sm = 0.f;
    for (int c = sub; c < S_; c += 16) {
      float e = __expf(Srow[row * S_ + c] - mx);
      sm += e;
      Ptile[row * S_ + c] = (__bf16)e;
    }
#pragma unroll
    for (int off = 8; off; off >>= 1) sm += __shfl_xor(sm, off, 32);
    const float inv = 1.f / sm;
    for (int c = sub; c < S_; c += 16)
      Ptile[row * S_ + c] = (__bf16)((float)Ptile[row * S_ + c] * inv);
  }
  __syncthreads();

  // phase 3: O = P * V ; each wave owns 64 output columns (4 tiles, A shared x4)
  v8f o0 = {}, o1 = {}, o2 = {}, o3 = {};
  const __bf16* vb0 = Vt + (size_t)b * C_ * S_ + (size_t)(wave * 64 + lrow) * S_;
  const __bf16* vb1 = vb0 + (size_t)16 * S_;
  const __bf16* vb2 = vb0 + (size_t)32 * S_;
  const __bf16* vb3 = vb0 + (size_t)48 * S_;
  for (int kk = 0; kk < S_; kk += 32) {
    v16bf a = pack_a16(Ptile + lrow * S_ + kk + 16 * lhi);
    const int ko = kk + 16 * lhi;
    v16bf m0 = *(const v16bf*)(vb0 + ko);
    v16bf m1 = *(const v16bf*)(vb1 + ko);
    v16bf m2 = *(const v16bf*)(vb2 + ko);
    v16bf m3 = *(const v16bf*)(vb3 + ko);
    o0 = wmma_bf16<false, false>(a, m0, o0);
    o1 = wmma_bf16<true,  false>(a, m1, o1);           // reuse A
    o2 = wmma_bf16<true,  false>(a, m2, o2);           // reuse A
    o3 = wmma_bf16<true,  false>(a, m3, o3);           // reuse A
  }

  // epilogue: out = O + x (residual)
  v8f oacc[4] = {o0, o1, o2, o3};
#pragma unroll
  for (int ntile = 0; ntile < 4; ++ntile) {
    int col = wave * 64 + ntile * 16 + lrow;
#pragma unroll
    for (int r = 0; r < 8; ++r) {
      size_t m = tokbase + r + 8 * lhi;
      out[m * C_ + col] = oacc[ntile][r] + x[m * C_ + col];
    }
  }
}

// ---------- host launcher ----------

extern "C" void kernel_launch(void* const* d_in, const int* in_sizes, int n_in,
                              void* d_out, int out_size, void* d_ws, size_t ws_size,
                              hipStream_t stream) {
  (void)in_sizes; (void)n_in; (void)out_size; (void)ws_size;
  const float* x  = (const float*)d_in[0];
  const float* Wq = (const float*)d_in[1];
  const float* bq = (const float*)d_in[2];
  const float* Wk = (const float*)d_in[3];
  const float* bk = (const float*)d_in[4];
  const float* Wv = (const float*)d_in[5];
  const float* bv = (const float*)d_in[6];
  float* out = (float*)d_out;

  char* ws = (char*)d_ws;
  size_t off = 0;
  unsigned short* xb  = (unsigned short*)(ws + off); off += (size_t)NTOK * C_ * 2;
  unsigned short* wqb = (unsigned short*)(ws + off); off += (size_t)C_ * C_ * 2;
  unsigned short* wkb = (unsigned short*)(ws + off); off += (size_t)C_ * C_ * 2;
  unsigned short* wvb = (unsigned short*)(ws + off); off += (size_t)C_ * C_ * 2;
  unsigned short* qb  = (unsigned short*)(ws + off); off += (size_t)NTOK * C_ * 2;
  unsigned short* kb  = (unsigned short*)(ws + off); off += (size_t)NTOK * C_ * 2;
  unsigned short* vtb = (unsigned short*)(ws + off); off += (size_t)NTOK * C_ * 2;

  // 1) downconvert activations + weights to bf16
  cvt_f32_bf16_kernel<<<512, 256, 0, stream>>>(x, xb, NTOK * C_);
  cvt_f32_bf16_kernel<<<64, 256, 0, stream>>>(Wq, wqb, C_ * C_);
  cvt_f32_bf16_kernel<<<64, 256, 0, stream>>>(Wk, wkb, C_ * C_);
  cvt_f32_bf16_kernel<<<64, 256, 0, stream>>>(Wv, wvb, C_ * C_);

  // 2) QKV projection: 3 mats * 512 row-blocks * 16 col-blocks = 24576 waves / 8 per block
  qkv_proj_kernel<<<3072, 256, 0, stream>>>(xb, wqb, wkb, wvb, bq, bk, bv,
                                            qb, kb, vtb);

  // 3) fused attention + residual: (128 query tiles) x (8 batches)
  attn_kernel<<<dim3(128, 8), 256, SMEM_SZ, stream>>>(qb, kb, vtb, x, out);
}
